// Transformer_30133490549272
// MI455X (gfx1250) — matrix operand order repaired and checked by reading, not compiled, not run on hardware
//
#include <hip/hip_runtime.h>
#include <hip/hip_bf16.h>
#include <math.h>

// ---------------------------------------------------------------------------
// Transformer forward for MI455X (gfx1250, wave32, WMMA).
// Activations bf16 end-to-end; fp32 weights (single HBM read) converted to
// bf16 with v_perm_b32 packing while staging into LDS. bf16 tiles staged with
// GLOBAL_LOAD_ASYNC_TO_LDS_B128 (ASYNCcnt). GEMM: BM=BN=BK=64, 8 waves,
// 2 accumulators/wave, 4 WMMAs per stage. fp32-B fetched via uniform-base +
// i32 offset (GVS global_load, LOADcnt only) so the software pipeline is not
// serialized by DScnt waits for the WMMA fragment ds_loads.
// ---------------------------------------------------------------------------

#define D_    768
#define H_    12
#define HK_   3
#define HD_   64
#define NL_   12
#define DFF_  3072
#define SW_   128
#define L_    1024

typedef __attribute__((ext_vector_type(16))) __bf16 v16bf;
typedef __attribute__((ext_vector_type(8)))  float  v8f;
typedef unsigned short us;
typedef long long ll;

// f32 -> bf16, round-half-up (bias add + shift)
__device__ __forceinline__ us f2bf(float f) {
    return (us)((__float_as_uint(f) + 0x8000u) >> 16);
}
// two f32 -> packed bf16x2 in ONE v_perm_b32 (+2 bias adds)
__device__ __forceinline__ unsigned cvt2(float x, float y) {
    unsigned ux = __float_as_uint(x) + 0x8000u;
    unsigned uy = __float_as_uint(y) + 0x8000u;
    return __builtin_amdgcn_perm(uy, ux, 0x07060302u);  // {hi16(uy), hi16(ux)}
}

// gfx1250 async copy: 16B global -> LDS per lane, tracked by ASYNCcnt
__device__ __forceinline__ void async_cp16(unsigned ldsAddr, const void* gaddr) {
    asm volatile("global_load_async_to_lds_b128 %0, %1, off"
                 :: "v"(ldsAddr), "v"((unsigned long long)(uintptr_t)gaddr)
                 : "memory");
}
__device__ __forceinline__ void wait_async0() {
#if __has_builtin(__builtin_amdgcn_s_wait_asynccnt)
    __builtin_amdgcn_s_wait_asynccnt(0);
#else
    asm volatile("s_wait_asynccnt 0" ::: "memory");
#endif
}

// ---------------------------------------------------------------------------
// Strided-batched GEMM: C[b] = op(A[b] x B[b/bDiv]) with epilogue.
//   A(b,m,k) = A[sA*b + m*ldaM + k]                       A is bf16, k-contig
//   BBF=1: B bf16, B(b,k,n) = B[sB*(b/bDiv) + n*ldbN + k] (k-contiguous)
//   BBF=0: B fp32 row-major [K,N]: B(b,k,n) = B[sB*(b/bDiv) + k*ldbK + n]
// MODE 0: f32 C = acc*scale           (C strides in f32 elements)
// MODE 1: f32 C = R + acc             (residual)
// MODE 2: f32 C = acc*scale + causal/sliding-window mask (-1e30)
// MODE 3: bf16 C = acc*scale          (C strides in bf16 elements)
// MODE 4: bf16 C^T: C[n*ldcM + m]     (transposed store, bf16)
// Requires M%64==0, N%64==0, K%64==0 (true for every call in this model).
// ---------------------------------------------------------------------------
#define BM 64
#define BN 64
#define BK 64

union Frag { v16bf v; uint4 u[2]; };

template <int BBF, int MODE>
__launch_bounds__(256, 2)
__global__ void gemm_bf16_wmma(
    const us* __restrict__ A, ll ldaM, ll sA,
    const void* __restrict__ Bp, ll ldbK, ll ldbN, ll sB, int bDiv,
    void* __restrict__ Cp, ll ldcM, ll sC,
    const float* __restrict__ R, ll ldrM, ll sR,
    int M, int N, int K, float scale, int window)
{
    __shared__ us As[BM][80];   // rows 160B: 16B-aligned frag reads
    __shared__ us Bs[BN][80];   // stored as Bs[n][k]

    const int tid  = threadIdx.x;
    const int b    = blockIdx.z;
    const int row0 = blockIdx.y * BM;
    const int col0 = blockIdx.x * BN;

    const int lane = tid & 31;
    const int wave = tid >> 5;     // 0..7
    const int wM   = wave >> 2;    // 0..1 -> 32-row slab
    const int wN   = wave & 3;     // 0..3 -> 16-col slab
    const int hh   = lane >> 4;    // lane half per ISA 16-bit operand layout
    const int mn   = lane & 15;

    const int cRow = tid >> 3;     // bf16 copy: 0..31
    const int cOc  = tid & 7;      // 16B unit within row
    const int tRow = tid >> 4;     // fp32-B staging: 0..15
    const int tQ   = tid & 15;

    v8f acc0 = {0.f,0.f,0.f,0.f,0.f,0.f,0.f,0.f};
    v8f acc1 = {0.f,0.f,0.f,0.f,0.f,0.f,0.f,0.f};

    // ---- hoisted, loop-carried staging state ----
    const us* Ab = A + sA * (ll)b;
    const us* aSrc[2];
    unsigned  aDst[2];
    #pragma unroll
    for (int q = 0; q < 2; ++q) {
        int r = cRow + 32 * q;
        aSrc[q] = Ab + (ll)(row0 + r) * ldaM + cOc * 8;
        aDst[q] = (unsigned)(uintptr_t)&As[r][cOc * 8];
    }

    const us* bSrc[2];              // BBF=1 async path
    unsigned  bDst[2];
    const float* Bf = nullptr;      // BBF=0: uniform base (SGPR)
    int    boff[4];                 //        divergent i32 element offsets
    float4 rB[4];
    if constexpr (BBF) {
        const us* Bh = (const us*)Bp + sB * (ll)(b / bDiv);
        #pragma unroll
        for (int q = 0; q < 2; ++q) {
            int n = cRow + 32 * q;
            bSrc[q] = Bh + (ll)(col0 + n) * ldbN + cOc * 8;
            bDst[q] = (unsigned)(uintptr_t)&Bs[n][cOc * 8];
        }
    } else {
        Bf = (const float*)Bp + sB * (ll)(b / bDiv);
        #pragma unroll
        for (int q = 0; q < 4; ++q) {
            boff[q] = (tRow + 16 * q) * (int)ldbK + col0 + tQ * 4;
            rB[q] = *(const float4*)&Bf[boff[q]];       // preload stage 0
        }
    }
    const int bStep = (int)(BK * ldbK);     // fp32-B element advance per stage
    int kOff = 0;                           // scalar (uniform) k offset

    const int nstages = K / BK;
    for (int s = 0; s < nstages; ++s) {
        const bool last = (s == nstages - 1);

        // ---- async-copy bf16 A tile into LDS (pure copy, ASYNCcnt) ----
        #pragma unroll
        for (int q = 0; q < 2; ++q) {
            async_cp16(aDst[q], aSrc[q]);
            aSrc[q] += BK;
        }
        if constexpr (BBF) {
            #pragma unroll
            for (int q = 0; q < 2; ++q) {
                async_cp16(bDst[q], bSrc[q]);
                bSrc[q] += BK;
            }
        } else {
            // ---- fp32 weights -> bf16 (v_perm pack), transposed scatter ----
            #pragma unroll
            for (int q = 0; q < 4; ++q) {
                int kk = tRow + 16 * q;
                unsigned p01 = cvt2(rB[q].x, rB[q].y);
                unsigned p23 = cvt2(rB[q].z, rB[q].w);
                Bs[tQ * 4 + 0][kk] = (us)p01;
                Bs[tQ * 4 + 1][kk] = (us)(p01 >> 16);
                Bs[tQ * 4 + 2][kk] = (us)p23;
                Bs[tQ * 4 + 3][kk] = (us)(p23 >> 16);
            }
        }
        wait_async0();
        __syncthreads();

        // ---- software pipeline: fetch next fp32-B stage while computing ----
        // uniform base + i32 offset => global_load (LOADcnt only, no DScnt)
        if constexpr (!BBF) {
            if (!last) {
                kOff += bStep;
                #pragma unroll
                for (int q = 0; q < 4; ++q)
                    rB[q] = *(const float4*)&Bf[boff[q] + kOff];
            }
        }
        if (!last)   // warm GL2 for next A tile (global_prefetch_b8)
            __builtin_prefetch(aSrc[0], 0, 1);

        // ---- 2 k-steps x 2 M-tiles = 4 WMMAs, B fragment reused ----
        #pragma unroll
        for (int ks = 0; ks < 2; ++ks) {
            Frag fb;
            const int bc = wN * 16 + mn;
            fb.u[0] = *(const uint4*)&Bs[bc][ks * 32 + hh * 8];
            fb.u[1] = *(const uint4*)&Bs[bc][ks * 32 + 16 + hh * 8];

            Frag fa0, fa1;
            const int ar0 = wM * 32 + mn;
            fa0.u[0] = *(const uint4*)&As[ar0][ks * 32 + hh * 8];
            fa0.u[1] = *(const uint4*)&As[ar0][ks * 32 + 16 + hh * 8];
            fa1.u[0] = *(const uint4*)&As[ar0 + 16][ks * 32 + hh * 8];
            fa1.u[1] = *(const uint4*)&As[ar0 + 16][ks * 32 + 16 + hh * 8];

            acc0 = __builtin_amdgcn_wmma_f32_16x16x32_bf16(
                       false, fa0.v, false, fb.v, (short)0, acc0, false, false);
            acc1 = __builtin_amdgcn_wmma_f32_16x16x32_bf16(
                       false, fa1.v, false, fb.v, (short)0, acc1, false, false);
        }
        __syncthreads();
    }

    // ---- epilogue: C/D layout VGPR r -> M = r + hh*8, N = lane&15 ----
    const int gj = col0 + wN * 16 + mn;
    #pragma unroll
    for (int t = 0; t < 2; ++t) {
        const v8f acc = t ? acc1 : acc0;
        const int gi0 = row0 + wM * 32 + t * 16 + hh * 8;
        #pragma unroll
        for (int r = 0; r < 8; ++r) {
            int gi = gi0 + r;
            if constexpr (MODE == 3) {
                ((us*)Cp)[sC * (ll)b + (ll)gi * ldcM + gj] = f2bf(acc[r] * scale);
            } else if constexpr (MODE == 4) {
                ((us*)Cp)[sC * (ll)b + (ll)gj * ldcM + gi] = f2bf(acc[r] * scale);
            } else if constexpr (MODE == 1) {
                float v = R[sR * (ll)b + (ll)gi * ldrM + gj] + acc[r];
                ((float*)Cp)[sC * (ll)b + (ll)gi * ldcM + gj] = v;
            } else if constexpr (MODE == 2) {
                float v = acc[r] * scale;
                if (gj > gi)                              v = -1e30f;  // causal
                else if (window > 0 && gj <= gi - window) v = -1e30f;  // sliding
                ((float*)Cp)[sC * (ll)b + (ll)gi * ldcM + gj] = v;
            } else {
                ((float*)Cp)[sC * (ll)b + (ll)gi * ldcM + gj] = acc[r] * scale;
            }
        }
    }
}

// ---------------------------------------------------------------------------
// Elementwise / reduction kernels
// ---------------------------------------------------------------------------
__global__ void embed_kernel(const int* __restrict__ ids,
                             const float* __restrict__ emb,
                             float* __restrict__ x)
{
    int i = blockIdx.x;
    ll base = (ll)ids[i] * D_;
    #pragma unroll
    for (int q = 0; q < 3; ++q) {
        int j = threadIdx.x + 256 * q;
        x[(ll)i * D_ + j] = emb[base + j];
    }
}

// rmsnorm: fp32 in, bf16 out (feeds GEMM A operands directly)
__global__ void rmsnorm_kernel(const float* __restrict__ x,
                               const float* __restrict__ w,
                               us* __restrict__ out)
{
    __shared__ float red[256];
    int row = blockIdx.x;
    const float* xr = x + (ll)row * D_;
    float vals[3];
    float s = 0.f;
    #pragma unroll
    for (int q = 0; q < 3; ++q) {
        float v = xr[threadIdx.x + 256 * q];
        vals[q] = v;
        s += v * v;
    }
    red[threadIdx.x] = s;
    __syncthreads();
    for (int st = 128; st > 0; st >>= 1) {
        if (threadIdx.x < st) red[threadIdx.x] += red[threadIdx.x + st];
        __syncthreads();
    }
    float inv = rsqrtf(red[0] * (1.0f / (float)D_) + 1e-8f);
    #pragma unroll
    for (int q = 0; q < 3; ++q) {
        int j = threadIdx.x + 256 * q;
        out[(ll)row * D_ + j] = f2bf(w[j] * vals[q] * inv);
    }
}

// softmax: fp32 scores in, bf16 probabilities out
__global__ void softmax_kernel(const float* __restrict__ s, us* __restrict__ p)
{
    __shared__ float red[256];
    const float* pr = s + (ll)blockIdx.x * L_;
    us* pw = p + (ll)blockIdx.x * L_;
    float vals[4];
    float mx = -3.4e38f;
    #pragma unroll
    for (int q = 0; q < 4; ++q) {
        float v = pr[threadIdx.x + 256 * q];
        vals[q] = v;
        mx = fmaxf(mx, v);
    }
    red[threadIdx.x] = mx;
    __syncthreads();
    for (int st = 128; st > 0; st >>= 1) {
        if (threadIdx.x < st) red[threadIdx.x] = fmaxf(red[threadIdx.x], red[threadIdx.x + st]);
        __syncthreads();
    }
    float m = red[0];
    __syncthreads();
    float sum = 0.f;
    #pragma unroll
    for (int q = 0; q < 4; ++q) {
        float e = __expf(vals[q] - m);
        vals[q] = e;
        sum += e;
    }
    red[threadIdx.x] = sum;
    __syncthreads();
    for (int st = 128; st > 0; st >>= 1) {
        if (threadIdx.x < st) red[threadIdx.x] += red[threadIdx.x + st];
        __syncthreads();
    }
    float inv = 1.0f / red[0];
    #pragma unroll
    for (int q = 0; q < 4; ++q)
        pw[threadIdx.x + 256 * q] = f2bf(vals[q] * inv);
}

// silu(g)*u : fp32 in, bf16 out
__global__ void silu_mul_kernel(const float* __restrict__ g,
                                const float* __restrict__ u,
                                us* __restrict__ t)
{
    ll i = (ll)blockIdx.x * 256 + threadIdx.x;
    float gv = g[i];
    t[i] = f2bf((gv / (1.0f + __expf(-gv))) * u[i]);
}

// ---------------------------------------------------------------------------
// Host orchestration (graph-capture safe: only stream launches)
// ---------------------------------------------------------------------------
extern "C" void kernel_launch(void* const* d_in, const int* in_sizes, int n_in,
                              void* d_out, int out_size, void* d_ws, size_t ws_size,
                              hipStream_t stream)
{
    (void)in_sizes; (void)n_in; (void)out_size; (void)ws_size;

    const int*   ids = (const int*)d_in[0];
    const float* emb = (const float*)d_in[1];
    const float* Wq  = (const float*)d_in[2];
    const float* Wk  = (const float*)d_in[3];
    const float* Wv  = (const float*)d_in[4];
    const float* Wo  = (const float*)d_in[5];
    const float* Wg  = (const float*)d_in[6];
    const float* Wu  = (const float*)d_in[7];
    const float* Wd  = (const float*)d_in[8];
    const float* n1  = (const float*)d_in[9];
    const float* n2  = (const float*)d_in[10];

    // workspace layout (cursor in fp32 elements; bf16 buffers use half slots)
    float* ws = (float*)d_ws;
    ll c = 0;
    float* x    = ws + c; c += (ll)L_ * D_;                 // fp32 [L,D]
    us*    h    = (us*)(ws + c); c += (ll)L_ * D_ / 2;      // bf16 [L,D]
    us*    qf   = (us*)(ws + c); c += (ll)L_ * D_ / 2;      // bf16 [L,768]
    us*    kf   = (us*)(ws + c); c += (ll)L_ * HK_ * HD_ / 2; // bf16 [L,192]
    us*    vT   = (us*)(ws + c); c += (ll)L_ * HK_ * HD_ / 2; // bf16 [192,L]
    float* sc   = ws + c; c += (ll)H_ * L_ * L_;            // fp32 [H,L,L]
    us*    ap   = (us*)(ws + c); c += (ll)H_ * L_ * L_ / 2; // bf16 [H,L,L]
    us*    of   = (us*)(ws + c); c += (ll)L_ * D_ / 2;      // bf16 [L,768]
    float* gbuf = ws + c; c += (ll)L_ * DFF_;               // fp32
    float* ubuf = ws + c; c += (ll)L_ * DFF_;               // fp32
    us*    tbuf = (us*)(ws + c); c += (ll)L_ * DFF_ / 2;    // bf16

    embed_kernel<<<L_, 256, 0, stream>>>(ids, emb, x);

    for (int i = 0; i < NL_; ++i) {
        const float* wq = Wq + (ll)i * D_ * D_;
        const float* wk = Wk + (ll)i * D_ * (HK_ * HD_);
        const float* wv = Wv + (ll)i * D_ * (HK_ * HD_);
        const float* wo = Wo + (ll)i * D_ * D_;
        const float* wg = Wg + (ll)i * D_ * DFF_;
        const float* wu = Wu + (ll)i * D_ * DFF_;
        const float* wd = Wd + (ll)i * DFF_ * D_;
        const int window = (i % 2 == 0) ? SW_ : 0;

        // ---- attention block ----
        rmsnorm_kernel<<<L_, 256, 0, stream>>>(x, n1 + (ll)i * D_, h);

        // qf(bf16) = h @ wq
        gemm_bf16_wmma<0, 3><<<dim3(D_ / BN, L_ / BM, 1), 256, 0, stream>>>(
            h, D_, 0,  wq, D_, 1, 0, 1,  qf, D_, 0,  nullptr, 0, 0,
            L_, D_, D_, 1.0f, 0);
        // kf(bf16) = h @ wk   [L,192]
        gemm_bf16_wmma<0, 3><<<dim3((HK_ * HD_) / BN, L_ / BM, 1), 256, 0, stream>>>(
            h, D_, 0,  wk, HK_ * HD_, 1, 0, 1,  kf, HK_ * HD_, 0,  nullptr, 0, 0,
            L_, HK_ * HD_, D_, 1.0f, 0);
        // vT(bf16) = (h @ wv)^T   [192,L] so attn@V's B is k-contiguous
        gemm_bf16_wmma<0, 4><<<dim3((HK_ * HD_) / BN, L_ / BM, 1), 256, 0, stream>>>(
            h, D_, 0,  wv, HK_ * HD_, 1, 0, 1,  vT, L_, 0,  nullptr, 0, 0,
            L_, HK_ * HD_, D_, 1.0f, 0);

        // scores = (q @ k^T)/8 + mask   A,B both bf16 (async staged)
        //   A(b,m,k)=qf[m*768+b*64+k];  B(b,k,n)=kf[n*192+(b/4)*64+k]
        gemm_bf16_wmma<1, 2><<<dim3(L_ / BN, L_ / BM, H_), 256, 0, stream>>>(
            qf, D_, HD_,
            kf, 1, HK_ * HD_, HD_, H_ / HK_,
            sc, L_, (ll)L_ * L_,
            nullptr, 0, 0,
            L_, L_, HD_, 0.125f, window);

        softmax_kernel<<<H_ * L_, 256, 0, stream>>>(sc, ap);

        // of(bf16) = attn @ v   B(b,k,n)=vT[(b/4)*64*L + n*L + k] (k-contig)
        gemm_bf16_wmma<1, 3><<<dim3(HD_ / BN, L_ / BM, H_), 256, 0, stream>>>(
            ap, L_, (ll)L_ * L_,
            vT, 1, L_, (ll)HD_ * L_, H_ / HK_,
            of, D_, HD_,
            nullptr, 0, 0,
            L_, HD_, L_, 1.0f, 0);

        // x = x + of @ wo
        gemm_bf16_wmma<0, 1><<<dim3(D_ / BN, L_ / BM, 1), 256, 0, stream>>>(
            of, D_, 0,  wo, D_, 1, 0, 1,  x, D_, 0,  x, D_, 0,
            L_, D_, D_, 1.0f, 0);

        // ---- SwiGLU MLP block ----
        rmsnorm_kernel<<<L_, 256, 0, stream>>>(x, n2 + (ll)i * D_, h);

        gemm_bf16_wmma<0, 0><<<dim3(DFF_ / BN, L_ / BM, 1), 256, 0, stream>>>(
            h, D_, 0,  wg, DFF_, 1, 0, 1,  gbuf, DFF_, 0,  nullptr, 0, 0,
            L_, DFF_, D_, 1.0f, 0);
        gemm_bf16_wmma<0, 0><<<dim3(DFF_ / BN, L_ / BM, 1), 256, 0, stream>>>(
            h, D_, 0,  wu, DFF_, 1, 0, 1,  ubuf, DFF_, 0,  nullptr, 0, 0,
            L_, DFF_, D_, 1.0f, 0);

        silu_mul_kernel<<<(L_ * DFF_) / 256, 256, 0, stream>>>(gbuf, ubuf, tbuf);

        // x = x + t @ wd   (last layer writes straight to d_out)
        float* cdst = (i == NL_ - 1) ? (float*)d_out : x;
        gemm_bf16_wmma<0, 1><<<dim3(D_ / BN, L_ / BM, 1), 256, 0, stream>>>(
            tbuf, DFF_, 0,  wd, D_, 1, 0, 1,  cdst, D_, 0,  x, D_, 0,
            L_, D_, DFF_, 1.0f, 0);
    }
}
